// CrossAttention_51814485459255
// MI455X (gfx1250) — compile-verified, matrix operand order checked by reading, not run on hardware
//
#include <hip/hip_runtime.h>

typedef __attribute__((ext_vector_type(16))) _Float16 h16;
typedef __attribute__((ext_vector_type(8)))  _Float16 h8;
typedef __attribute__((ext_vector_type(8)))  float    f8;
typedef __attribute__((ext_vector_type(4)))  unsigned int v4u;
typedef __attribute__((ext_vector_type(8)))  int v8i;
typedef __attribute__((ext_vector_type(4)))  int v4i;

#define WMMA(a,b,c) __builtin_amdgcn_wmma_f32_16x16x32_f16(false,(a),false,(b),(short)0,(c),false,false)

// Build a 16-half A/B fragment from two contiguous 8-half runs in LDS.
// CDNA5 16-bit A 16x32 layout: lane L holds row L%16; K runs
// [koff, koff+8) and [16+koff, 16+koff+8) with koff = (L>=16)*8.
__device__ __forceinline__ h16 ldfrag(const _Float16* lo, const _Float16* hi) {
  h8 a = *(const h8*)lo;
  h8 b = *(const h8*)hi;
  return __builtin_shufflevector(a, b, 0,1,2,3,4,5,6,7,8,9,10,11,12,13,14,15);
}

__device__ __forceinline__ unsigned pack2f(float a, float b) {
  union { _Float16 h[2]; unsigned u; } t;
  t.h[0] = (_Float16)a; t.h[1] = (_Float16)b;
  return t.u;
}

// TDM: DMA a 2D f16 tile [rows x 64] (row stride 64 elems in memory) into LDS,
// padding LDS rows to 80 halves (pad 8 DWORDs after every 32 DWORDs) so the
// destination matches the PAD=80 LDS tiling used by the WMMA fragment loads.
__device__ __forceinline__ void tdm_load_2d(unsigned lds_addr, const _Float16* gptr,
                                            int rows, int tensor_rows) {
  unsigned long long ga = (unsigned long long)(size_t)gptr;
  v4u g0;
  g0[0] = 1u;                                   // count=1, user descriptor
  g0[1] = lds_addr;                             // LDS byte address
  g0[2] = (unsigned)ga;                         // global_addr[31:0]
  g0[3] = (unsigned)(ga >> 32) | (2u << 30);    // global_addr[56:32] | type=2
  v8i g1;
  g1[0] = (1 << 16)        // data_size = 2 bytes
        | (1 << 20)        // pad_enable
        | (4 << 22)        // pad_interval: after 32 DWORDs (one 64-half row)
        | (7 << 25);       // pad_amount: 8 DWORDs (16 halves) -> 80-half pitch
  g1[1] = (64 << 16);          // tensor_dim0 = 64 (bits 63:48)
  g1[2] = (tensor_rows << 16); // tensor_dim1[15:0] (bits 95:80)
  g1[3] = (64 << 16);          // tile_dim0 = 64 (bits 127:112)
  g1[4] = rows;                // tile_dim1 (bits 143:128), tile_dim2 = 0
  g1[5] = 64;                  // tensor_dim0_stride = 64 (bits 191:160)
  g1[6] = 0;
  g1[7] = 0;
  v4i z4 = {0, 0, 0, 0};               // groups 2/3 unused (<=2D tensor)
  v8i z8 = {0, 0, 0, 0, 0, 0, 0, 0};   // trailing group (clang-23 6-arg form)
  __builtin_amdgcn_tensor_load_to_lds(g0, g1, z4, z4, z8, 0);
}

// C = A[M,K] * W[Nout,K]^T + bias.  MODE 0: write f16 Q layout [B,H,seq,64].
// MODE 1: split cols into K/V f16 [B,H,seq,64]. MODE 2: write fp32 row-major.
template<int MODE, bool AF32>
__global__ __launch_bounds__(256)
void gemm_wmma_kernel(const void* __restrict__ Ap, const float* __restrict__ W,
                      const float* __restrict__ bias,
                      float* __restrict__ oF32, _Float16* __restrict__ oA,
                      _Float16* __restrict__ oB,
                      int M, int K, int Nout, int seqlen)
{
  constexpr int BM = 128, BN = 64, BK = 32, PAD = 40;
  __shared__ _Float16 As[BM * PAD];
  __shared__ _Float16 Bs[BN * PAD];
  const int tid = threadIdx.x, lane = tid & 31, wave = tid >> 5;
  const int wm = wave & 3, wn = wave >> 2;          // 4x2 wave grid, 32x32 each
  const int rl = lane & 15, koff = (lane >> 4) * 8;
  const int bm = blockIdx.y * BM, bn = blockIdx.x * BN;

  f8 acc[2][2] = {};

  for (int k0 = 0; k0 < K; k0 += BK) {
    if constexpr (AF32) {
      const float* A = (const float*)Ap;
      #pragma unroll
      for (int i = 0; i < 4; ++i) {
        int c = tid + 256 * i;
        int row = c >> 3, col = (c & 7) * 4;
        float4 f = *(const float4*)(A + (size_t)(bm + row) * K + k0 + col);
        uint2 st; st.x = pack2f(f.x, f.y); st.y = pack2f(f.z, f.w);
        *(uint2*)&As[row * PAD + col] = st;
      }
      if (k0 + BK < K)   // prefetch next K-step of A into cache
        __builtin_prefetch(A + (size_t)(bm + (tid >> 3)) * K + k0 + BK + (tid & 7) * 4, 0, 1);
    } else {
      const _Float16* A = (const _Float16*)Ap;
      #pragma unroll
      for (int i = 0; i < 2; ++i) {
        int c = tid + 256 * i;
        int row = c >> 2, col = (c & 3) * 8;
        *(h8*)&As[row * PAD + col] = *(const h8*)(A + (size_t)(bm + row) * K + k0 + col);
      }
      if (k0 + BK < K)
        __builtin_prefetch(A + (size_t)(bm + (tid >> 2)) * K + k0 + BK + (tid & 3) * 8, 0, 1);
    }
    #pragma unroll
    for (int i = 0; i < 2; ++i) {
      int c = tid + 256 * i;
      int row = c >> 3, col = (c & 7) * 4;
      float4 f = *(const float4*)(W + (size_t)(bn + row) * K + k0 + col);
      uint2 st; st.x = pack2f(f.x, f.y); st.y = pack2f(f.z, f.w);
      *(uint2*)&Bs[row * PAD + col] = st;
    }
    __syncthreads();

    h16 af[2], bf[2];
    #pragma unroll
    for (int s = 0; s < 2; ++s) {
      const _Float16* p = &As[(wm * 32 + s * 16 + rl) * PAD];
      af[s] = ldfrag(p + koff, p + 16 + koff);
    }
    #pragma unroll
    for (int s = 0; s < 2; ++s) {
      const _Float16* p = &Bs[(wn * 32 + s * 16 + rl) * PAD];
      bf[s] = ldfrag(p + koff, p + 16 + koff);
    }
    #pragma unroll
    for (int sm = 0; sm < 2; ++sm)
      #pragma unroll
      for (int sn = 0; sn < 2; ++sn)
        acc[sm][sn] = WMMA(af[sm], bf[sn], acc[sm][sn]);
    __syncthreads();
  }

  // C/D layout: VGPR r -> M = r + 8*(lane>=16), N = lane%16 within 16x16 tile.
  const int mh = (lane >> 4) << 3;
  #pragma unroll
  for (int sm = 0; sm < 2; ++sm)
  #pragma unroll
  for (int sn = 0; sn < 2; ++sn)
  #pragma unroll
  for (int r = 0; r < 8; ++r) {
    int row = bm + wm * 32 + sm * 16 + r + mh;
    int col = bn + wn * 32 + sn * 16 + rl;
    float v = acc[sm][sn][r] + bias[col];
    if constexpr (MODE == 2) {
      oF32[(size_t)row * Nout + col] = v;
    } else if constexpr (MODE == 0) {
      int b = row / seqlen, n = row % seqlen;
      int h = col >> 6, hd = col & 63;
      oA[(((size_t)b * 8 + h) * seqlen + n) * 64 + hd] = (_Float16)v;
    } else {
      int b = row / seqlen, s = row % seqlen;
      _Float16* dst = oA;
      int c = col;
      if (c >= 512) { c -= 512; dst = oB; }
      int h = c >> 6, hd = c & 63;
      dst[(((size_t)b * 8 + h) * seqlen + s) * 64 + hd] = (_Float16)v;
    }
  }
}

// Flash cross-attention: one WG = 64 Q rows of one (b,h); 4 waves x 16 rows.
// Q and K tiles arrive via the Tensor Data Mover (async DMA into padded LDS);
// V is loaded+transposed manually. Online softmax; P@V via WMMA.
__global__ __launch_bounds__(128)
void flash_attn_kernel(const _Float16* __restrict__ Q, const _Float16* __restrict__ Kk,
                       const _Float16* __restrict__ Vv, const unsigned char* __restrict__ mask,
                       _Float16* __restrict__ AO)
{
  constexpr int N = 4096, S = 512, PAD = 80;
  constexpr float SCALE = 0.125f;  // 64^-0.5
  __shared__ _Float16 Qs[64 * PAD];
  __shared__ _Float16 Ks[64 * PAD];     // [s][hd], filled by TDM
  __shared__ _Float16 Vt[64 * PAD];     // transposed: [hd][s]
  __shared__ _Float16 Ps[4][16 * PAD];  // per-wave P scratch

  const int tid = threadIdx.x, lane = tid & 31, wave = tid >> 5;
  const int rl = lane & 15, koff = (lane >> 4) * 8;
  const int b = blockIdx.z, h = blockIdx.y;
  const int bh = b * 8 + h;
  const int n0 = blockIdx.x * 64;

  const _Float16* Qg = Q + ((size_t)bh * N + n0) * 64;
  const _Float16* Kg = Kk + (size_t)bh * S * 64;
  const _Float16* Vg = Vv + (size_t)bh * S * 64;
  const unsigned char* mk = mask + (size_t)b * S;

  const unsigned qs_lds = (unsigned)(size_t)&Qs[0];
  const unsigned ks_lds = (unsigned)(size_t)&Ks[0];

  // async Q tile load via TDM (wave 0 issues; TENSORcnt is per-wave)
  if (wave == 0) {
    tdm_load_2d(qs_lds, Qg, 64, 64);
    __builtin_amdgcn_s_wait_tensorcnt(0);
  }
  __syncthreads();

  h16 qf[2];
  {
    const _Float16* p = &Qs[(wave * 16 + rl) * PAD];
    qf[0] = ldfrag(p + koff, p + 16 + koff);        // hd 0..31
    qf[1] = ldfrag(p + 32 + koff, p + 48 + koff);   // hd 32..63
  }

  float m[8], l[8];
  f8 o[4] = {};
  #pragma unroll
  for (int r = 0; r < 8; ++r) { m[r] = -1e30f; l[r] = 0.f; }

  for (int s0 = 0; s0 < S; s0 += 64) {
    __syncthreads();
    // K tile via TDM (overlaps with the manual V transpose-fill below)
    if (wave == 0) tdm_load_2d(ks_lds, Kg + (size_t)s0 * 64, 64, 64);
    #pragma unroll
    for (int i = 0; i < 4; ++i) {
      int c = tid + 128 * i;
      int s = c >> 3, hd0 = (c & 7) * 8;
      h8 v = *(const h8*)(Vg + (s0 + s) * 64 + hd0);
      #pragma unroll
      for (int j = 0; j < 8; ++j) Vt[(hd0 + j) * PAD + s] = v[j];
    }
    if (wave == 0) __builtin_amdgcn_s_wait_tensorcnt(0);
    __syncthreads();

    // S_ij = Q K^T for 4 column tiles of 16
    f8 sf[4];
    #pragma unroll
    for (int t = 0; t < 4; ++t) {
      const _Float16* p = &Ks[(t * 16 + rl) * PAD];
      h16 kf0 = ldfrag(p + koff, p + 16 + koff);
      h16 kf1 = ldfrag(p + 32 + koff, p + 48 + koff);
      f8 z = {};
      z = WMMA(qf[0], kf0, z);
      z = WMMA(qf[1], kf1, z);
      sf[t] = z;
    }
    #pragma unroll
    for (int t = 0; t < 4; ++t) {
      bool pad = mk[s0 + t * 16 + rl] != 0;   // True = padding -> -inf
      #pragma unroll
      for (int r = 0; r < 8; ++r)
        sf[t][r] = pad ? -1e30f : sf[t][r] * SCALE;
    }
    // online softmax; rows live across 16-lane groups
    #pragma unroll
    for (int r = 0; r < 8; ++r) {
      float v = fmaxf(fmaxf(sf[0][r], sf[1][r]), fmaxf(sf[2][r], sf[3][r]));
      #pragma unroll
      for (int d = 1; d < 16; d <<= 1) v = fmaxf(v, __shfl_xor(v, d, 32));
      float mn = fmaxf(m[r], v);
      float corr = __expf(m[r] - mn);
      float rs = 0.f;
      #pragma unroll
      for (int t = 0; t < 4; ++t) { float p = __expf(sf[t][r] - mn); sf[t][r] = p; rs += p; }
      #pragma unroll
      for (int d = 1; d < 16; d <<= 1) rs += __shfl_xor(rs, d, 32);
      l[r] = l[r] * corr + rs;
      m[r] = mn;
      #pragma unroll
      for (int t = 0; t < 4; ++t) o[t][r] *= corr;
    }
    // C-layout -> A-layout reshape of P through per-wave LDS
    _Float16* Pw = Ps[wave];
    const int mh = (lane >> 4) << 3;
    #pragma unroll
    for (int t = 0; t < 4; ++t)
      #pragma unroll
      for (int r = 0; r < 8; ++r)
        Pw[(r + mh) * PAD + t * 16 + rl] = (_Float16)sf[t][r];
    asm volatile("s_wait_dscnt 0x0" ::: "memory");  // per-wave store->load on LDS
    h16 pf0, pf1;
    {
      const _Float16* p = &Pw[rl * PAD];
      pf0 = ldfrag(p + koff, p + 16 + koff);        // s' 0..31
      pf1 = ldfrag(p + 32 + koff, p + 48 + koff);   // s' 32..63
    }
    #pragma unroll
    for (int t = 0; t < 4; ++t) {
      const _Float16* p = &Vt[(t * 16 + rl) * PAD];
      h16 vf0 = ldfrag(p + koff, p + 16 + koff);
      h16 vf1 = ldfrag(p + 32 + koff, p + 48 + koff);
      o[t] = WMMA(pf0, vf0, o[t]);
      o[t] = WMMA(pf1, vf1, o[t]);
    }
  }

  const int mh2 = (lane >> 4) << 3;
  #pragma unroll
  for (int t = 0; t < 4; ++t)
    #pragma unroll
    for (int r = 0; r < 8; ++r) {
      float v = o[t][r] / l[r];
      int n = n0 + wave * 16 + r + mh2;
      int hd = t * 16 + rl;
      AO[((size_t)b * N + n) * 512 + h * 64 + hd] = (_Float16)v;
    }
}

extern "C" void kernel_launch(void* const* d_in, const int* in_sizes, int n_in,
                              void* d_out, int out_size, void* d_ws, size_t ws_size,
                              hipStream_t stream)
{
  (void)in_sizes; (void)n_in; (void)out_size; (void)ws_size;
  const float* x    = (const float*)d_in[0];
  const float* ctx  = (const float*)d_in[1];
  const unsigned char* mask = (const unsigned char*)d_in[2];
  const float* wq   = (const float*)d_in[3];
  const float* bq   = (const float*)d_in[4];
  const float* wkv  = (const float*)d_in[5];
  const float* bkv  = (const float*)d_in[6];
  const float* wp   = (const float*)d_in[7];
  const float* bp   = (const float*)d_in[8];
  float* out = (float*)d_out;

  char* ws = (char*)d_ws;
  const size_t QBYTES = (size_t)8 * 8 * 4096 * 64 * 2;  // 33,554,432
  const size_t KBYTES = (size_t)8 * 8 * 512  * 64 * 2;  //  4,194,304
  _Float16* Qws  = (_Float16*)ws;
  _Float16* Kws  = (_Float16*)(ws + QBYTES);
  _Float16* Vws  = (_Float16*)(ws + QBYTES + KBYTES);
  _Float16* AOws = (_Float16*)(ws + QBYTES + 2 * KBYTES);

  // Q = x @ wq^T + bq  -> f16 [B,H,N,64]
  gemm_wmma_kernel<0, true><<<dim3(512 / 64, 32768 / 128), 256, 0, stream>>>(
      x, wq, bq, nullptr, Qws, nullptr, 32768, 512, 512, 4096);
  // KV = ctx @ wkv^T + bkv -> K,V f16 [B,H,S,64]
  gemm_wmma_kernel<1, true><<<dim3(1024 / 64, 4096 / 128), 256, 0, stream>>>(
      ctx, wkv, bkv, nullptr, Kws, Vws, 4096, 512, 1024, 512);
  // fused masked-softmax attention -> f16 [B,N,512]
  flash_attn_kernel<<<dim3(4096 / 64, 8, 8), 128, 0, stream>>>(Qws, Kws, Vws, mask, AOws);
  // out = AO @ wp^T + bp -> fp32
  gemm_wmma_kernel<2, false><<<dim3(512 / 64, 32768 / 128), 256, 0, stream>>>(
      AOws, wp, bp, out, nullptr, nullptr, 32768, 512, 512, 4096);
}
